// GlobalAttentionFeatureEncoder_82437602280032
// MI455X (gfx1250) — matrix operand order repaired and checked by reading, not compiled, not run on hardware
//
#include <hip/hip_runtime.h>
#include <math.h>

// ---------------- problem constants ----------------
#define B_    2
#define CIN_  256
#define CV_   128
#define HID_  512
#define H_    96
#define W_    320
#define WH_   12           // window height (H_/8)
#define WW_   40           // window width  (W_/8)
#define HW_   (H_*W_)
#define NP_   (B_*HW_)     // 61440 tokens
#define WT_   (WH_*WW_)    // 480 tokens / window
#define NWIN_ (B_*64)      // 128 windows total

typedef __attribute__((ext_vector_type(16))) __bf16 v16bf;
typedef __attribute__((ext_vector_type(8)))  float  v8f;

// pixel index (b,h,w) -> window-token row index
__device__ __forceinline__ int pix2win_bhw(int b, int h, int w) {
  int win = (h / WH_) * 8 + (w / WW_);
  int t   = (h % WH_) * WW_ + (w % WW_);
  return b * HW_ + win * WT_ + t;
}
__device__ __forceinline__ int pix2win_m(int m) {
  int b = m / HW_, r = m % HW_;
  return pix2win_bhw(b, r / W_, r % W_);
}
// C-row = true pixel; A-row = window row of rolled coords
__device__ __forceinline__ int mapA_row(int m, int shh, int shw) {
  int b = m / HW_, r = m % HW_;
  int h = r / W_, w = r % W_;
  int hr = h - shh; if (hr < 0) hr += H_;
  int wr = w - shw; if (wr < 0) wr += W_;
  return pix2win_bhw(b, hr, wr);
}

__device__ __forceinline__ int swin_region(int h, int w) {
  int rh = (h < H_ - WH_) ? 0 : ((h < H_ - WH_ / 2) ? 1 : 2);
  int rw = (w < W_ - WW_) ? 0 : ((w < W_ - WW_ / 2) ? 1 : 2);
  return rh * 3 + rw;
}
__device__ __forceinline__ float swin_mask_val(int z, int q, int k) {
  int win = z & 63;
  int wy = win >> 3, wx = win & 7;
  int qh = wy * WH_ + q / WW_, qw = wx * WW_ + q % WW_;
  int kh = wy * WH_ + k / WW_, kw = wx * WW_ + k % WW_;
  return (swin_region(qh, qw) != swin_region(kh, kw)) ? -100.0f : 0.0f;
}

// ---------------- generic batched WMMA GEMM ----------------
// C[m,n] = alpha * sum_k bf16(A[m,k]) * bf16(B[k,n])  (+bias +resid, relu, swin-mask)
// REQUIREMENT: K % 32 == 0 and 16B-aligned rows (true for every call site here).
// Out-of-range tile rows/cols are CLAMPED (not predicated); their products land in
// accumulator lanes that the epilogue never stores.
// transB: B stored (N,K) row-major.  permC: scatter C rows pixel->window.
// permA: gather A rows via window-row of (pixel rolled by -shh,-shw).
struct GemmP {
  const float* A; const float* B; const float* bias; const float* resid; float* C;
  int M, N, K, lda, ldb, ldc;
  long long sA, sB, sC;
  float alpha;
  int transB, relu, swinmask, permC, permA, shh, shw;
};

#define BM 64
#define BN 128
#define BK 32
#define LDK (BK + 8)   // padded k-stride (avoids LDS bank conflicts, keeps 16B align)

__global__ void __launch_bounds__(256) k_gemm(GemmP p) {
  __shared__ __align__(16) __bf16 As[2][BM][LDK];  // [m][k]
  __shared__ __align__(16) __bf16 Bs[2][BN][LDK];  // [n][k] (transposed stage)

  const int bz = blockIdx.z;
  const float* Ab = p.A + (long long)bz * p.sA;
  const float* Bb = p.B + (long long)bz * p.sB;
  float*       Cb = p.C + (long long)bz * p.sC;
  const float* Rb = p.resid ? p.resid + (long long)bz * p.sC : nullptr;

  const int m0blk = blockIdx.y * BM;
  const int n0blk = blockIdx.x * BN;
  const int tid  = threadIdx.x;
  const int lane = tid & 31, wid = tid >> 5;
  const int wm = wid >> 2, wn = wid & 3;      // 2 x 4 wave grid
  const int half = lane >> 4, l15 = lane & 15;

  // ---- k-invariant staging setup (indices clamped; no load predication) ----
  const int  sr  = tid >> 2;             // A: row in tile (0..63)
  const int  skq = (tid & 3) * 8;        // A: k offset (0/8/16/24)
  int sgm = m0blk + sr; if (sgm > p.M - 1) sgm = p.M - 1;
  const long long arow = (long long)(p.permA ? mapA_row(sgm, p.shh, p.shw) : sgm);
  const float* aptr = Ab + arow * p.lda + skq;

  const int  sn  = tid & 127;            // B: col in tile (0..127)
  const int  skh = (tid >> 7) * 16;      // B: k offset (0/16)
  int sgn = n0blk + sn; if (sgn > p.N - 1) sgn = p.N - 1;
  const float* bptrT = Bb + (long long)sgn * p.ldb + skh;   // transB base
  const float* bptrN = Bb + (long long)skh * p.ldb + sgn;   // normal base

  auto stageA = [&](int k0, int buf) {
    float4 x0 = *(const float4*)(aptr + k0);
    float4 x1 = *(const float4*)(aptr + k0 + 4);
    __bf16* d = &As[buf][sr][skq];
    d[0] = (__bf16)x0.x; d[1] = (__bf16)x0.y; d[2] = (__bf16)x0.z; d[3] = (__bf16)x0.w;
    d[4] = (__bf16)x1.x; d[5] = (__bf16)x1.y; d[6] = (__bf16)x1.z; d[7] = (__bf16)x1.w;
  };
  auto stageB = [&](int k0, int buf) {
    float v[16];
    if (p.transB) {
      const float4* src = (const float4*)(bptrT + k0);
      float4 u0 = src[0], u1 = src[1], u2 = src[2], u3 = src[3];
      v[0]=u0.x; v[1]=u0.y; v[2]=u0.z; v[3]=u0.w;
      v[4]=u1.x; v[5]=u1.y; v[6]=u1.z; v[7]=u1.w;
      v[8]=u2.x; v[9]=u2.y; v[10]=u2.z; v[11]=u2.w;
      v[12]=u3.x; v[13]=u3.y; v[14]=u3.z; v[15]=u3.w;
    } else {
      const float* src = bptrN + (long long)k0 * p.ldb;
      #pragma unroll
      for (int i = 0; i < 16; i++) v[i] = src[(long long)i * p.ldb];
    }
    __bf16* d = &Bs[buf][sn][skh];
    #pragma unroll
    for (int i = 0; i < 16; i++) d[i] = (__bf16)v[i];
  };

  v8f acc[2][2] = {};

  const int nT = p.K / BK;
  stageA(0, 0);
  stageB(0, 0);

  for (int t = 0; t < nT; t++) {
    __syncthreads();
    if (t + 2 < nT) __builtin_prefetch(aptr + (t + 2) * BK, 0, 1);
    if (t + 1 < nT) {                         // overlap next tile's loads with WMMA
      stageA((t + 1) * BK, (t + 1) & 1);
      stageB((t + 1) * BK, (t + 1) & 1);
    }
    const int buf = t & 1;

    // ---- fragments (ISA 16-bit A 16x32 / B 32x16 layouts) ----
    union U { v16bf v; uint4 u[2]; };
    U fa[2], fb[2];
    #pragma unroll
    for (int i = 0; i < 2; i++) {
      const uint4* pa = (const uint4*)&As[buf][wm * 32 + i * 16 + l15][0];
      fa[i].u[0] = pa[half];          // k = 8*half .. +8
      fa[i].u[1] = pa[2 + half];      // k = 16+8*half .. +8
      const uint4* pb = (const uint4*)&Bs[buf][wn * 32 + i * 16 + l15][0];
      fb[i].u[0] = pb[half];          // k = 8*half .. +8
      fb[i].u[1] = pb[2 + half];      // k = 16+8*half .. +8
    }
    #pragma unroll
    for (int i = 0; i < 2; i++)
      #pragma unroll
      for (int j = 0; j < 2; j++)
        acc[i][j] = __builtin_amdgcn_wmma_f32_16x16x32_bf16(
            false, fa[i].v, false, fb[j].v, (short)0, acc[i][j], false, false);
  }

  // ---- epilogue ----
  #pragma unroll
  for (int i = 0; i < 2; i++) {
    #pragma unroll
    for (int r = 0; r < 8; r++) {
      int m = m0blk + wm * 32 + i * 16 + half * 8 + r;
      if (m >= p.M) continue;
      long long mo = p.permC ? (long long)pix2win_m(m) : (long long)m;
      #pragma unroll
      for (int j = 0; j < 2; j++) {
        int n = n0blk + wn * 32 + j * 16 + l15;
        if (n >= p.N) continue;
        float v = p.alpha * acc[i][j][r];
        if (p.bias) v += p.bias[n];
        if (Rb)     v += Rb[(long long)m * p.ldc + n];
        if (p.relu) v = fmaxf(v, 0.f);
        if (p.swinmask) v += swin_mask_val(bz, m, n);
        Cb[mo * p.ldc + n] = v;
      }
    }
  }
}

// ---------------- layout transposes ----------------
__global__ void k_nchw2nhwc(const float* __restrict__ x, float* __restrict__ xa,
                            int C, long long total) {
  long long idx = (long long)blockIdx.x * 256 + threadIdx.x;
  if (idx >= total) return;
  int w = (int)(idx % W_); long long t = idx / W_;
  int h = (int)(t % H_);   t /= H_;
  int c = (int)(t % C);    int b = (int)(t / C);
  long long pix = (long long)b * HW_ + (long long)h * W_ + w;
  xa[pix * C + c] = x[idx];
}
__global__ void k_nhwc2nchw(const float* __restrict__ f, float* __restrict__ o,
                            long long total) {
  long long idx = (long long)blockIdx.x * 256 + threadIdx.x;
  if (idx >= total) return;
  int w = (int)(idx % W_); long long t = idx / W_;
  int h = (int)(t % H_);   t /= H_;
  int c = (int)(t % CV_);  int b = (int)(t / CV_);
  long long pix = (long long)b * HW_ + (long long)h * W_ + w;
  o[idx] = f[pix * CV_ + c];
}

// ---------------- LayerNorm (optional roll by (-shh,-shw)) ----------------
__global__ void k_layernorm(const float* __restrict__ f, const float* __restrict__ g,
                            const float* __restrict__ b, float* __restrict__ out,
                            int shh, int shw) {
  __shared__ float red[128];
  int row = blockIdx.x;
  int tid = threadIdx.x;
  int bb = row / HW_, r = row % HW_;
  int h = r / W_, w = r % W_;
  int hs = h + shh; if (hs >= H_) hs -= H_;
  int ws = w + shw; if (ws >= W_) ws -= W_;
  long long src = (((long long)bb * H_ + hs) * W_ + ws) * CV_ + tid;
  float v = f[src];
  red[tid] = v; __syncthreads();
  for (int s = 64; s > 0; s >>= 1) { if (tid < s) red[tid] += red[tid + s]; __syncthreads(); }
  float mu = red[0] * (1.0f / CV_); __syncthreads();
  float d = v - mu;
  red[tid] = d * d; __syncthreads();
  for (int s = 64; s > 0; s >>= 1) { if (tid < s) red[tid] += red[tid + s]; __syncthreads(); }
  float var = red[0] * (1.0f / CV_);
  out[(long long)row * CV_ + tid] = d * rsqrtf(var + 1e-5f) * g[tid] + b[tid];
}

// ---------------- sink softmax, 9-wide (outlook) ----------------
__global__ void k_softmax9(float* __restrict__ a, long long rows) {
  long long row = (long long)blockIdx.x * 256 + threadIdx.x;
  if (row >= rows) return;
  float* p = a + row * 9;
  float v[9], m = 0.f;
  #pragma unroll
  for (int i = 0; i < 9; i++) { v[i] = p[i]; m = fmaxf(m, v[i]); }
  float s = expf(-m);
  #pragma unroll
  for (int i = 0; i < 9; i++) { v[i] = expf(v[i] - m); s += v[i]; }
  float inv = 1.f / s;
  #pragma unroll
  for (int i = 0; i < 9; i++) p[i] = v[i] * inv;
}

// ---------------- sink softmax, 480-wide (window attn), in-place ----------------
__global__ void k_softmax480(float* __restrict__ S) {
  __shared__ float red[256];
  long long row = blockIdx.x;
  float* base = S + row * WT_;
  int tid = threadIdx.x;
  float mx = -1e30f;
  for (int i = tid; i < WT_; i += 256) mx = fmaxf(mx, base[i]);
  red[tid] = mx; __syncthreads();
  for (int s = 128; s > 0; s >>= 1) { if (tid < s) red[tid] = fmaxf(red[tid], red[tid + s]); __syncthreads(); }
  float m = fmaxf(red[0], 0.f); __syncthreads();
  float sm = 0.f;
  for (int i = tid; i < WT_; i += 256) { float e = expf(base[i] - m); base[i] = e; sm += e; }
  red[tid] = sm; __syncthreads();
  for (int s = 128; s > 0; s >>= 1) { if (tid < s) red[tid] += red[tid + s]; __syncthreads(); }
  float inv = 1.f / (red[0] + expf(-m)); __syncthreads();
  for (int i = tid; i < WT_; i += 256) base[i] *= inv;
}

// ---------------- outlook attention gather ----------------
// y[n,c] = sum_{p=(i,j),q=(qi,qj)} A[nbr(i,j)][p*9+q] * v[h+qi-i, w+qj-j, c]
__global__ void k_outlook_gather(const float* __restrict__ attn,
                                 const float* __restrict__ v,
                                 float* __restrict__ y) {
  __shared__ float sA[81];
  int pxl = blockIdx.x;
  int tid = threadIdx.x;
  int b = pxl / HW_, r = pxl % HW_;
  int h = r / W_, w = r % W_;
  if (tid < 81) {
    int p = tid / 9;
    int i = p / 3, j = p % 3;
    int hp = h + 1 - i, wp = w + 1 - j;
    float a = 0.f;
    if (hp >= 0 && hp < H_ && wp >= 0 && wp < W_)
      a = attn[(((long long)b * H_ + hp) * W_ + wp) * 81 + tid];
    sA[tid] = a;
  }
  __syncthreads();
  int c = tid;
  float vv[5][5];
  #pragma unroll
  for (int dy = -2; dy <= 2; dy++)
    #pragma unroll
    for (int dx = -2; dx <= 2; dx++) {
      int hh = h + dy, ww = w + dx;
      vv[dy + 2][dx + 2] = (hh >= 0 && hh < H_ && ww >= 0 && ww < W_)
          ? v[(((long long)b * H_ + hh) * W_ + ww) * CV_ + c] : 0.f;
    }
  float acc = 0.f;
  #pragma unroll
  for (int i = 0; i < 3; i++)
    #pragma unroll
    for (int j = 0; j < 3; j++)
      #pragma unroll
      for (int qi = 0; qi < 3; qi++)
        #pragma unroll
        for (int qj = 0; qj < 3; qj++)
          acc += sA[(i * 3 + j) * 9 + qi * 3 + qj] * vv[qi - i + 2][qj - j + 2];
  y[(long long)pxl * CV_ + c] = acc;
}

// ---------------- depthwise 3x3 + bias + exact GELU (NHWC, C=512) ----------------
__global__ void k_dwconv_gelu(const float* __restrict__ in, const float* __restrict__ dw,
                              const float* __restrict__ bdw, float* __restrict__ out) {
  long long idx = (long long)blockIdx.x * 256 + threadIdx.x;
  if (idx >= (long long)NP_ * HID_) return;
  int c = (int)(idx & (HID_ - 1));
  long long pxl = idx >> 9;
  int b = (int)(pxl / HW_); int r = (int)(pxl % HW_);
  int h = r / W_, w = r % W_;
  float s = 0.f;
  #pragma unroll
  for (int dy = -1; dy <= 1; dy++)
    #pragma unroll
    for (int dx = -1; dx <= 1; dx++) {
      int hh = h + dy, ww = w + dx;
      if (hh >= 0 && hh < H_ && ww >= 0 && ww < W_)
        s += in[(((long long)b * H_ + hh) * W_ + ww) * HID_ + c]
           * dw[c * 9 + (dy + 1) * 3 + (dx + 1)];
    }
  s += bdw[c];
  out[idx] = 0.5f * s * (1.f + erff(s * 0.70710678118654752f));
}

// ---------------- sinusoidal positional embedding add ----------------
__global__ void k_posadd(float* __restrict__ f) {
  long long idx = (long long)blockIdx.x * 256 + threadIdx.x;
  if (idx >= (long long)NP_ * CV_) return;
  int c = (int)(idx & (CV_ - 1));
  long long pxl = idx >> 7;
  int r = (int)(pxl % HW_);
  int h = r / W_, w = r % W_;
  const float TWO_PI = 6.283185307179586f;
  float val;
  if (c < 64) {
    int m = c >> 1;
    float p = (float)(h + 1) * (TWO_PI / ((float)H_ + 1e-6f));
    float t = p / powf(10000.f, (float)m * (1.f / 32.f));
    val = (c & 1) ? cosf(t) : sinf(t);
  } else {
    int cc = c - 64; int m = cc >> 1;
    float p = (float)(w + 1) * (TWO_PI / ((float)W_ + 1e-6f));
    float t = p / powf(10000.f, (float)m * (1.f / 32.f));
    val = (cc & 1) ? cosf(t) : sinf(t);
  }
  f[idx] += val;
}

// ---------------- host-side helpers ----------------
static inline void gemm(hipStream_t s, const float* A, const float* B, const float* bias,
                        const float* resid, float* C, int M, int N, int K,
                        int lda, int ldb, int ldc,
                        long long sA, long long sB, long long sC, int batch,
                        float alpha, int transB, int relu, int swin,
                        int permC, int permA, int shh, int shw) {
  GemmP p{A, B, bias, resid, C, M, N, K, lda, ldb, ldc, sA, sB, sC,
          alpha, transB, relu, swin, permC, permA, shh, shw};
  dim3 g((N + BN - 1) / BN, (M + BM - 1) / BM, batch), blk(256);
  k_gemm<<<g, blk, 0, s>>>(p);
}

static inline void conv_ffn_block(hipStream_t s, float* f, float* xn, float* t0, float* t1,
                                  const float* g, const float* b, const float* w1,
                                  const float* b1, const float* dw, const float* bdw,
                                  const float* w2, const float* b2) {
  k_layernorm<<<NP_, 128, 0, s>>>(f, g, b, xn, 0, 0);
  gemm(s, xn, w1, b1, nullptr, t0, NP_, HID_, CV_, CV_, HID_, HID_,
       0, 0, 0, 1, 1.f, 0, 0, 0, 0, 0, 0, 0);
  k_dwconv_gelu<<<(int)(((long long)NP_ * HID_ + 255) / 256), 256, 0, s>>>(t0, dw, bdw, t1);
  gemm(s, t1, w2, b2, f, f, NP_, CV_, HID_, HID_, CV_, CV_,
       0, 0, 0, 1, 1.f, 0, 0, 0, 0, 0, 0, 0);
}

static inline void window_block(hipStream_t s, float* f, float* xn, float* scores, float* aout,
                                float* qw, float* kw, float* vw,
                                const float* g, const float* b, const float* wq,
                                const float* wk, const float* wv, const float* wo,
                                const float* bo, int shift) {
  int sh = shift ? WH_ / 2 : 0, sw = shift ? WW_ / 2 : 0;
  k_layernorm<<<NP_, 128, 0, s>>>(f, g, b, xn, sh, sw);
  // projections, scattered straight into window-token layout (permC)
  gemm(s, xn, wq, nullptr, nullptr, qw, NP_, CV_, CV_, CV_, CV_, CV_,
       0, 0, 0, 1, 1.f, 0, 0, 0, 1, 0, 0, 0);
  gemm(s, xn, wk, nullptr, nullptr, kw, NP_, CV_, CV_, CV_, CV_, CV_,
       0, 0, 0, 1, 1.f, 0, 0, 0, 1, 0, 0, 0);
  gemm(s, xn, wv, nullptr, nullptr, vw, NP_, CV_, CV_, CV_, CV_, CV_,
       0, 0, 0, 1, 1.f, 0, 0, 0, 1, 0, 0, 0);
  // scores = Q K^T / sqrt(C) (+ swin mask), batched over 128 windows
  gemm(s, qw, kw, nullptr, nullptr, scores, WT_, WT_, CV_, CV_, CV_, WT_,
       (long long)WT_ * CV_, (long long)WT_ * CV_, (long long)WT_ * WT_, NWIN_,
       0.08838834764831845f, 1, 0, shift ? 1 : 0, 0, 0, 0, 0);
  k_softmax480<<<NWIN_ * WT_, 256, 0, s>>>(scores);
  // out = P V
  gemm(s, scores, vw, nullptr, nullptr, aout, WT_, CV_, WT_, WT_, CV_, CV_,
       (long long)WT_ * WT_, (long long)WT_ * CV_, (long long)WT_ * CV_, NWIN_,
       1.f, 0, 0, 0, 0, 0, 0, 0);
  // output projection: gather A rows from window layout (with un-roll), + residual
  gemm(s, aout, wo, bo, f, f, NP_, CV_, CV_, CV_, CV_, CV_,
       0, 0, 0, 1, 1.f, 0, 0, 0, 0, 1, sh, sw);
}

// ---------------- entry point ----------------
extern "C" void kernel_launch(void* const* d_in, const int* in_sizes, int n_in,
                              void* d_out, int out_size, void* d_ws, size_t ws_size,
                              hipStream_t stream) {
  (void)in_sizes; (void)n_in; (void)out_size; (void)ws_size;
  const float* x = (const float*)d_in[0];
  const float *agg_w1 = (const float*)d_in[1],  *agg_b1 = (const float*)d_in[2];
  const float *agg_w2 = (const float*)d_in[3],  *agg_b2 = (const float*)d_in[4];
  const float *ol_g = (const float*)d_in[5],  *ol_b = (const float*)d_in[6];
  const float *ol_wv = (const float*)d_in[7], *ol_wa = (const float*)d_in[8];
  const float *ol_ba = (const float*)d_in[9], *ol_wo = (const float*)d_in[10];
  const float *ol_bo = (const float*)d_in[11];
  const float *f1_g = (const float*)d_in[12], *f1_b = (const float*)d_in[13];
  const float *f1_w1 = (const float*)d_in[14], *f1_b1 = (const float*)d_in[15];
  const float *f1_dw = (const float*)d_in[16], *f1_bdw = (const float*)d_in[17];
  const float *f1_w2 = (const float*)d_in[18], *f1_b2 = (const float*)d_in[19];
  const float *w_g = (const float*)d_in[20], *w_b = (const float*)d_in[21];
  const float *w_wq = (const float*)d_in[22], *w_wk = (const float*)d_in[23];
  const float *w_wv = (const float*)d_in[24], *w_wo = (const float*)d_in[25];
  const float *w_bo = (const float*)d_in[26];
  const float *f2_g = (const float*)d_in[27], *f2_b = (const float*)d_in[28];
  const float *f2_w1 = (const float*)d_in[29], *f2_b1 = (const float*)d_in[30];
  const float *f2_dw = (const float*)d_in[31], *f2_bdw = (const float*)d_in[32];
  const float *f2_w2 = (const float*)d_in[33], *f2_b2 = (const float*)d_in[34];
  const float *s_g = (const float*)d_in[35], *s_b = (const float*)d_in[36];
  const float *s_wq = (const float*)d_in[37], *s_wk = (const float*)d_in[38];
  const float *s_wv = (const float*)d_in[39], *s_wo = (const float*)d_in[40];
  const float *s_bo = (const float*)d_in[41];
  const float *f3_g = (const float*)d_in[42], *f3_b = (const float*)d_in[43];
  const float *f3_w1 = (const float*)d_in[44], *f3_b1 = (const float*)d_in[45];
  const float *f3_dw = (const float*)d_in[46], *f3_bdw = (const float*)d_in[47];
  const float *f3_w2 = (const float*)d_in[48], *f3_b2 = (const float*)d_in[49];

  // workspace layout (floats)
  float* ws   = (float*)d_ws;
  float* f    = ws;                               // NP*128
  float* xn   = f    + (long long)NP_ * CV_;      // NP*128
  float* big0 = xn   + (long long)NP_ * CV_;      // NP*512 (xa / ffn hid / ol logits / scores)
  float* big1 = big0 + (long long)NP_ * HID_;     // NP*512 (ffn hid2 / attn out)
  float* ra   = big1 + (long long)NP_ * HID_;     // NP*128 (ol v / q_win)
  float* rb   = ra   + (long long)NP_ * CV_;      // NP*128 (ol y / k_win)
  float* rc   = rb   + (long long)NP_ * CV_;      // NP*128 (v_win)
  float* xa   = big0;                             // NP*256 alias

  hipStream_t s = stream;

  // 1) NCHW -> NHWC, then agg MLP
  {
    long long tot = (long long)B_ * CIN_ * HW_;
    k_nchw2nhwc<<<(int)((tot + 255) / 256), 256, 0, s>>>(x, xa, CIN_, tot);
  }
  gemm(s, xa, agg_w1, agg_b1, nullptr, xn, NP_, 128, CIN_, CIN_, 128, 128,
       0, 0, 0, 1, 1.f, 0, /*relu*/1, 0, 0, 0, 0, 0);
  gemm(s, xn, agg_w2, agg_b2, nullptr, f, NP_, CV_, 128, 128, CV_, CV_,
       0, 0, 0, 1, 1.f, 0, 0, 0, 0, 0, 0, 0);

  // 2) outlook attention (residual)
  k_layernorm<<<NP_, 128, 0, s>>>(f, ol_g, ol_b, xn, 0, 0);
  gemm(s, xn, ol_wv, nullptr, nullptr, ra, NP_, CV_, CV_, CV_, CV_, CV_,
       0, 0, 0, 1, 1.f, 0, 0, 0, 0, 0, 0, 0);
  gemm(s, xn, ol_wa, ol_ba, nullptr, big0, NP_, 81, CV_, CV_, 81, 81,
       0, 0, 0, 1, 1.f, 0, 0, 0, 0, 0, 0, 0);
  k_softmax9<<<(int)(((long long)NP_ * 9 + 255) / 256), 256, 0, s>>>(big0, (long long)NP_ * 9);
  k_outlook_gather<<<NP_, 128, 0, s>>>(big0, ra, rb);
  gemm(s, rb, ol_wo, ol_bo, f, f, NP_, CV_, CV_, CV_, CV_, CV_,
       0, 0, 0, 1, 1.f, 0, 0, 0, 0, 0, 0, 0);

  // 3) conv-FFN 1
  conv_ffn_block(s, f, xn, big0, big1, f1_g, f1_b, f1_w1, f1_b1, f1_dw, f1_bdw, f1_w2, f1_b2);

  // 4) positional embedding
  k_posadd<<<(int)(((long long)NP_ * CV_ + 255) / 256), 256, 0, s>>>(f);

  // 5) window attention (no shift)
  window_block(s, f, xn, big0, big1, ra, rb, rc, w_g, w_b, w_wq, w_wk, w_wv, w_wo, w_bo, 0);

  // 6) conv-FFN 2
  conv_ffn_block(s, f, xn, big0, big1, f2_g, f2_b, f2_w1, f2_b1, f2_dw, f2_bdw, f2_w2, f2_b2);

  // 7) shifted window attention
  window_block(s, f, xn, big0, big1, ra, rb, rc, s_g, s_b, s_wq, s_wk, s_wv, s_wo, s_bo, 1);

  // 8) conv-FFN 3
  conv_ffn_block(s, f, xn, big0, big1, f3_g, f3_b, f3_w1, f3_b1, f3_dw, f3_bdw, f3_w2, f3_b2);

  // 9) NHWC -> NCHW output
  {
    long long tot = (long long)B_ * CV_ * HW_;
    k_nhwc2nchw<<<(int)((tot + 255) / 256), 256, 0, s>>>(f, (float*)d_out, tot);
  }
}
